// BiLSTMTagger_15857019256979
// MI455X (gfx1250) — compile-verified, matrix operand order checked by reading, not run on hardware
//
#include <hip/hip_runtime.h>
#include <hip/hip_bf16.h>

// ---------------------------------------------------------------------------
// BiLSTM tagger for MI455X (gfx1250, wave32, WMMA).
// embed-gather -> bf16 conversions -> per-layer [WMMA GEMM (pre), sequential
// LSTM (WMMA matvec)] -> head/dep GEMMs -> biaffine (2 GEMMs) -> row softmax.
// GEMM tiles are staged with CDNA5 async-to-LDS copies (ASYNCcnt), matrix
// math on v_wmma_f32_16x16x32_bf16 with f32 accumulation.
// ---------------------------------------------------------------------------

typedef __bf16 bf16;
typedef __attribute__((ext_vector_type(16))) __bf16 v16bf;
typedef __attribute__((ext_vector_type(8)))  __bf16 bf16x8;
typedef __attribute__((ext_vector_type(8)))  float  v8f;

#define S_LEN 1024
#define D_DIM 1024
#define H_DIM 512
#define G4H   2048   // 4*H (per direction gate width)
#define EW_W  896
#define EP_W  128

// --- LDS byte-offset of a generic pointer known to alias LDS ---------------
__device__ __forceinline__ unsigned lds_off_of(const void* p) {
  return (unsigned)reinterpret_cast<unsigned long long>(
      (__attribute__((address_space(3))) const void*)p);
}

// --- CDNA5 async copy: 16 bytes global -> LDS per lane (ASYNCcnt) ----------
__device__ __forceinline__ void async_copy_b128(unsigned lds_dst, const void* gsrc) {
  asm volatile("global_load_async_to_lds_b128 %0, %1, off"
               :: "v"(lds_dst), "v"((unsigned long long)gsrc)
               : "memory");
}
__device__ __forceinline__ void wait_async0() {
  asm volatile("s_wait_asynccnt 0x0" ::: "memory");
}

// --- fragment loader: 16 bf16 (two b128 loads) in ISA 16-bit A/B layout -----
// lane half (=lane>>4) selects which K sub-slices this lane owns:
//   lanes 0-15 : K = {0..7, 16..23},  lanes 16-31 : K = {8..15, 24..31}
__device__ __forceinline__ v16bf load_frag(const bf16* __restrict__ rowBase, int half) {
  bf16x8 lo = *(const bf16x8*)(rowBase + half * 8);
  bf16x8 hi = *(const bf16x8*)(rowBase + 16 + half * 8);
  return __builtin_shufflevector(lo, hi, 0,1,2,3,4,5,6,7,8,9,10,11,12,13,14,15);
}

// --- embedding gather + bf16 pack ------------------------------------------
__global__ void k_embed(const int* __restrict__ x, const int* __restrict__ xp,
                        const float* __restrict__ we, const float* __restrict__ pe,
                        bf16* __restrict__ Xb) {
  const int s = blockIdx.x;
  const int w = x[s], p = xp[s];
  for (int j = threadIdx.x; j < D_DIM; j += blockDim.x) {
    float v = (j < EW_W) ? we[(size_t)w * EW_W + j] : pe[(size_t)p * EP_W + (j - EW_W)];
    Xb[(size_t)s * D_DIM + j] = (bf16)v;
  }
}

// --- f32 -> bf16 ------------------------------------------------------------
__global__ void k_cvt_bf16(const float* __restrict__ in, bf16* __restrict__ out, int n) {
  int i = blockIdx.x * blockDim.x + threadIdx.x;
  if (i < n) out[i] = (bf16)in[i];
}

// --- TN GEMM: C[M,N] = A[M,K] * B[N,K]^T (+bias), bf16 in, f32 accumulate ---
// block tile 64x256, 8 waves (2x4), wave tile 32x64 (2x4 WMMA frags), K step 32
// Tiles staged via global_load_async_to_lds_b128 (ASYNCcnt).
#define TM 64
#define TN 256
#define TK 32

__global__ __launch_bounds__(256)
void k_gemm_tn(const bf16* __restrict__ A, const bf16* __restrict__ B,
               const float* __restrict__ bias, int bias_mode, // 0 none, 1 per-N, 2 scalar
               float* __restrict__ Cf, bf16* __restrict__ Cb,
               int M, int N, int K) {
  __shared__ __align__(16) bf16 Al[TM * TK];   //  4 KB
  __shared__ __align__(16) bf16 Bl[TN * TK];   // 16 KB
  const int tid   = threadIdx.x;
  const int lane  = tid & 31;
  const int wv    = tid >> 5;
  const int half  = lane >> 4;
  const int lm    = lane & 15;
  const int waveM = wv >> 2;       // 0..1
  const int waveN = wv & 3;        // 0..3
  const int mBase = blockIdx.y * TM;
  const int nBase = blockIdx.x * TN;

  // per-thread staging coordinates (16B granules)
  const int arow = tid >> 2, aco = (tid & 3) * 8;

  v8f acc[2][4] = {};

  for (int kc = 0; kc < K; kc += TK) {
    // stage A tile 64x32: one async b128 per thread
    async_copy_b128(lds_off_of(&Al[arow * TK + aco]),
                    &A[(size_t)(mBase + arow) * K + kc + aco]);
    // stage B tile 256x32: four async b128 per thread
    #pragma unroll
    for (int i = 0; i < 4; ++i) {
      const int idx = tid + i * 256;
      const int row = idx >> 2, co = (idx & 3) * 8;
      async_copy_b128(lds_off_of(&Bl[row * TK + co]),
                      &B[(size_t)(nBase + row) * K + kc + co]);
    }
    wait_async0();
    __syncthreads();

    v16bf af[2], bfr[4];
    #pragma unroll
    for (int mi = 0; mi < 2; ++mi)
      af[mi] = load_frag(&Al[(waveM * 32 + mi * 16 + lm) * TK], half);
    #pragma unroll
    for (int ni = 0; ni < 4; ++ni)
      bfr[ni] = load_frag(&Bl[(waveN * 64 + ni * 16 + lm) * TK], half);

    #pragma unroll
    for (int mi = 0; mi < 2; ++mi)
      #pragma unroll
      for (int ni = 0; ni < 4; ++ni)
        acc[mi][ni] = __builtin_amdgcn_wmma_f32_16x16x32_bf16(
            false, af[mi], false, bfr[ni], (short)0, acc[mi][ni], false, false);
    __syncthreads();
  }

  // epilogue: lane l, vgpr r -> (M = r + 8*half, N = lm)
  #pragma unroll
  for (int mi = 0; mi < 2; ++mi) {
    #pragma unroll
    for (int ni = 0; ni < 4; ++ni) {
      const int col = nBase + waveN * 64 + ni * 16 + lm;
      float bv = 0.f;
      if (bias_mode == 1) bv = bias[col];
      else if (bias_mode == 2) bv = bias[0];
      #pragma unroll
      for (int r = 0; r < 8; ++r) {
        const int row = mBase + waveM * 32 + mi * 16 + r + half * 8;
        float v = acc[mi][ni][r] + bv;
        if (Cf) Cf[(size_t)row * N + col] = v;
        if (Cb) Cb[(size_t)row * N + col] = (bf16)v;
      }
    }
  }
}

// --- sequential bidirectional LSTM layer ------------------------------------
// grid = 2 blocks (dir 0 fwd / dir 1 bwd), 1024 threads = 32 waves.
// Per step: g = W_hh * h via WMMA (A = W rows, B = h broadcast over N columns;
// D is replicated across N so lanes lm==0 extract it), then gate nonlinearity.
__global__ __launch_bounds__(1024)
void k_lstm_seq(const float* __restrict__ pre,   // [S, 4096] (dir0 gates | dir1 gates)
                const bf16* __restrict__ Whh,    // [2, 2048, 512] bf16 (this layer)
                bf16* __restrict__ outb) {       // [S, 1024] bf16 (fwd|bwd halves)
  const int d    = blockIdx.x;
  const int tid  = threadIdx.x;
  const int lane = tid & 31;
  const int wv   = tid >> 5;       // 0..31
  const int half = lane >> 4;
  const int lm   = lane & 15;

  __shared__ __align__(16) float g[G4H];
  __shared__ __align__(16) float c[H_DIM];
  __shared__ __align__(16) bf16  hb[H_DIM];

  for (int j = tid; j < H_DIM; j += 1024) { c[j] = 0.f; hb[j] = (bf16)0.f; }
  __syncthreads();

  const bf16*  W    = Whh + (size_t)d * G4H * H_DIM;
  const float* preD = pre + (size_t)d * G4H;

  for (int ts = 0; ts < S_LEN; ++ts) {
    const int t = d ? (S_LEN - 1 - ts) : ts;

    // ---- phase 1: g[0..2047] = W_hh @ h  (+ pre) ----
    #pragma unroll
    for (int q = 0; q < 4; ++q) {
      const int mt = wv * 4 + q;                 // 0..127 : 16-row tile of W
      v8f acc = {};
      const bf16* wrow = W + (size_t)(mt * 16 + lm) * H_DIM;
      for (int kc = 0; kc < H_DIM; kc += 32) {
        __builtin_prefetch(wrow + kc + 64, 0, 1);   // global_prefetch_b8 (L2-resident W)
        v16bf af  = load_frag(wrow + kc, half);     // A: 16 rows of W, K chunk
        v16bf bfr = load_frag(&hb[kc], half);       // B: h broadcast over columns
        acc = __builtin_amdgcn_wmma_f32_16x16x32_bf16(
            false, af, false, bfr, (short)0, acc, false, false);
      }
      if (lm == 0) {                             // lanes 0 & 16 own the result
        const int base = mt * 16 + half * 8;
        #pragma unroll
        for (int r = 0; r < 8; ++r)
          g[base + r] = acc[r] + preD[(size_t)t * 4096 + base + r];
      }
    }
    __syncthreads();

    // ---- phase 2: gates i,f,g,o -> c,h ----
    if (tid < H_DIM) {
      const int j = tid;
      float i_ = g[j], f_ = g[H_DIM + j], g_ = g[2 * H_DIM + j], o_ = g[3 * H_DIM + j];
      float si = 1.f / (1.f + __expf(-i_));
      float sf = 1.f / (1.f + __expf(-f_));
      float so = 1.f / (1.f + __expf(-o_));
      float cn = sf * c[j] + si * tanhf(g_);
      float hn = so * tanhf(cn);
      c[j]  = cn;
      hb[j] = (bf16)hn;
      outb[(size_t)t * D_DIM + d * H_DIM + j] = (bf16)hn;
    }
    __syncthreads();
  }
}

// --- in-place row softmax on [S, S] f32 -------------------------------------
__global__ __launch_bounds__(256)
void k_softmax(float* __restrict__ sc) {
  float* p = sc + (size_t)blockIdx.x * S_LEN;
  __shared__ float red[256];
  const int tid = threadIdx.x;
  float v[4];
  float m = -1e30f;
  #pragma unroll
  for (int i = 0; i < 4; ++i) { v[i] = p[tid + i * 256]; m = fmaxf(m, v[i]); }
  red[tid] = m; __syncthreads();
  for (int s = 128; s > 0; s >>= 1) {
    if (tid < s) red[tid] = fmaxf(red[tid], red[tid + s]);
    __syncthreads();
  }
  m = red[0]; __syncthreads();
  float sum = 0.f;
  #pragma unroll
  for (int i = 0; i < 4; ++i) { v[i] = __expf(v[i] - m); sum += v[i]; }
  red[tid] = sum; __syncthreads();
  for (int s = 128; s > 0; s >>= 1) {
    if (tid < s) red[tid] += red[tid + s];
    __syncthreads();
  }
  const float inv = 1.f / red[0];
  #pragma unroll
  for (int i = 0; i < 4; ++i) p[tid + i * 256] = v[i] * inv;
}

// ---------------------------------------------------------------------------
extern "C" void kernel_launch(void* const* d_in, const int* in_sizes, int n_in,
                              void* d_out, int out_size, void* d_ws, size_t ws_size,
                              hipStream_t stream) {
  const int*   x      = (const int*)  d_in[0];
  const int*   x_pos  = (const int*)  d_in[1];
  const float* we     = (const float*)d_in[2];
  const float* pe     = (const float*)d_in[3];
  const float* W_ih   = (const float*)d_in[4];   // [2,2,2048,1024]
  const float* W_hh   = (const float*)d_in[5];   // [2,2,2048,512]
  const float* b_lstm = (const float*)d_in[6];   // [2,2,2048]
  const float* Wh     = (const float*)d_in[7];
  const float* bh     = (const float*)d_in[8];
  const float* Wd     = (const float*)d_in[9];
  const float* bd     = (const float*)d_in[10];
  const float* Wbi    = (const float*)d_in[11];
  const float* bbi    = (const float*)d_in[12];
  float* scores = (float*)d_out;                 // [S, S]

  // ---- workspace layout (256B aligned bumps) ----
  char* base = (char*)d_ws;
  size_t off = 0;
  auto bump = [&](size_t bytes) -> char* {
    char* p = base + off;
    off += (bytes + 255) & ~(size_t)255;
    return p;
  };
  bf16*  actA  = (bf16*) bump((size_t)S_LEN * D_DIM * 2);        // layer io A
  bf16*  actB  = (bf16*) bump((size_t)S_LEN * D_DIM * 2);        // layer io B
  float* pre   = (float*)bump((size_t)S_LEN * 2 * G4H * 4);      // [S,4096]
  bf16*  Wihb  = (bf16*) bump((size_t)2 * 2 * G4H * D_DIM * 2);  // 16MB
  bf16*  Whhb  = (bf16*) bump((size_t)2 * 2 * G4H * H_DIM * 2);  // 8MB
  bf16*  Whb   = (bf16*) bump((size_t)D_DIM * D_DIM * 2);
  bf16*  Wdb   = (bf16*) bump((size_t)D_DIM * D_DIM * 2);
  bf16*  Wbib  = (bf16*) bump((size_t)D_DIM * D_DIM * 2);
  bf16*  headb = (bf16*) bump((size_t)S_LEN * D_DIM * 2);
  bf16*  depb  = (bf16*) bump((size_t)S_LEN * D_DIM * 2);
  bf16*  tmp2b = (bf16*) bump((size_t)S_LEN * D_DIM * 2);
  (void)ws_size; (void)n_in; (void)in_sizes; (void)out_size;

  // ---- 1. embedding gather ----
  k_embed<<<S_LEN, 256, 0, stream>>>(x, x_pos, we, pe, actA);

  // ---- 2. weight conversions to bf16 ----
  auto cvt = [&](const float* src, bf16* dst, int n) {
    k_cvt_bf16<<<(n + 255) / 256, 256, 0, stream>>>(src, dst, n);
  };
  cvt(W_ih, Wihb, 2 * 2 * G4H * D_DIM);
  cvt(W_hh, Whhb, 2 * 2 * G4H * H_DIM);
  cvt(Wh,   Whb,  D_DIM * D_DIM);
  cvt(Wd,   Wdb,  D_DIM * D_DIM);
  cvt(Wbi,  Wbib, D_DIM * D_DIM);

  // ---- 3. two stacked BiLSTM layers ----
  bf16* cur = actA;
  bf16* nxt = actB;
  for (int l = 0; l < 2; ++l) {
    // pre = X @ W_ih[l]^T + b_lstm[l]   (N covers both directions: 4096)
    dim3 gp(2 * G4H / TN, S_LEN / TM);
    k_gemm_tn<<<gp, 256, 0, stream>>>(
        cur, Wihb + (size_t)l * 2 * G4H * D_DIM,
        b_lstm + (size_t)l * 2 * G4H, /*bias_mode=*/1,
        pre, (bf16*)nullptr, S_LEN, 2 * G4H, D_DIM);
    // recurrence: 2 blocks (fwd, bwd)
    k_lstm_seq<<<2, 1024, 0, stream>>>(
        pre, Whhb + (size_t)l * 2 * G4H * H_DIM, nxt);
    bf16* t = cur; cur = nxt; nxt = t;
  }
  // `cur` now holds final BiLSTM output [S, 1024] in bf16

  // ---- 4. head / dep projections ----
  dim3 gq(D_DIM / TN, S_LEN / TM);
  k_gemm_tn<<<gq, 256, 0, stream>>>(cur, Whb, bh, 1,
                                    (float*)nullptr, headb, S_LEN, D_DIM, D_DIM);
  k_gemm_tn<<<gq, 256, 0, stream>>>(cur, Wdb, bd, 1,
                                    (float*)nullptr, depb, S_LEN, D_DIM, D_DIM);

  // ---- 5. biaffine: tmp2 = dep @ Wbi^T ; scores = head @ tmp2^T + bbi ----
  k_gemm_tn<<<gq, 256, 0, stream>>>(depb, Wbib, (const float*)nullptr, 0,
                                    (float*)nullptr, tmp2b, S_LEN, D_DIM, D_DIM);
  dim3 gs(S_LEN / TN, S_LEN / TM);
  k_gemm_tn<<<gs, 256, 0, stream>>>(headb, tmp2b, bbi, /*scalar*/2,
                                    scores, (bf16*)nullptr, S_LEN, S_LEN, D_DIM);

  // ---- 6. row softmax (in place on d_out) ----
  k_softmax<<<S_LEN, 256, 0, stream>>>(scores);
}
